// MoEDecoderLayer_17600775979497
// MI455X (gfx1250) — matrix operand order repaired and checked by reading, not compiled, not run on hardware
//
#include <hip/hip_runtime.h>
#include <hip/hip_bf16.h>

// ---------------------------------------------------------------------------
// MoE decoder layer for MI455X (gfx1250, wave32, WMMA bf16 path)
// B=16, S=1024, T=512, D=512, H=8, DF=2048, E=4
// GEMMs: bf16 operands (weights pre-transposed), double-buffered LDS tiles
// filled with GLOBAL_LOAD_ASYNC_TO_LDS_B128 (ASYNCcnt), software-pipelined
// with a 2x-unrolled steady state, fragments via ds_load_b128, math via
// v_wmma_f32_16x16x32_bf16 with f32 accumulation.
// ---------------------------------------------------------------------------

typedef __attribute__((ext_vector_type(16))) __bf16 v16bf;
typedef __attribute__((ext_vector_type(8)))  float  v8f;

#define B_  16
#define S_  1024
#define T_  512
#define D_  512
#define H_  8
#define DH_ 64
#define DF_ 2048
#define E_  4
#define BS_ (B_ * S_)
#define BT_ (B_ * T_)
#define EPS_ 1e-5f

__device__ __forceinline__ unsigned short bfbits(float f) {
    union { float f; unsigned u; } a; a.f = f;
    return (unsigned short)((a.u + 0x7FFFu + ((a.u >> 16) & 1u)) >> 16);
}
__device__ __forceinline__ __bf16 f2bf(float f) {
    return __builtin_bit_cast(__bf16, bfbits(f));
}
// pack two f32 -> packed bf16x2 (low = first arg)
__device__ __forceinline__ unsigned bfpack(float lo, float hi) {
#if __has_builtin(__builtin_amdgcn_cvt_pk_bf16_f32)
    typedef __attribute__((ext_vector_type(2))) __bf16 v2bf;
    v2bf p = __builtin_amdgcn_cvt_pk_bf16_f32(lo, hi);
    return __builtin_bit_cast(unsigned, p);
#else
    return (unsigned)bfbits(lo) | ((unsigned)bfbits(hi) << 16);
#endif
}

// A-fragment swizzle: position of k within a 32-wide row so that each lane's
// 16 fragment halves are contiguous.  p in [0,32)
__device__ __forceinline__ int aswz(int k) {
    return ((k & 8) << 1) + ((k & 16) >> 1) + (k & 7);
}

union Frag16 {
    struct { uint4 lo, hi; } q;
    v16bf v;
};
union FragU {           // build a fragment from 8 packed dwords
    unsigned u[8];
    v16bf v;
};

// issue one per-lane async global->LDS 16-byte copy (ASYNCcnt path)
__device__ __forceinline__ void async_b128(unsigned lds_off, const void* gptr) {
    const unsigned long long ga = (unsigned long long)(size_t)gptr;
    asm volatile("global_load_async_to_lds_b128 %0, %1, off"
                 :: "v"(lds_off), "v"(ga) : "memory");
}

// ===========================================================================
// One-time converters (bandwidth-bound, run once per launch)
// ===========================================================================

// row-major f32 -> row-major bf16, 8 elements / thread
__global__ __launch_bounds__(256) void convert_rm(
    const float* __restrict__ in, __bf16* __restrict__ out) {
    const size_t i8 = ((size_t)blockIdx.x * 256 + threadIdx.x) * 8;
    const float4 a = *(const float4*)(in + i8);
    const float4 b = *(const float4*)(in + i8 + 4);
    uint4 o;
    o.x = bfpack(a.x, a.y); o.y = bfpack(a.z, a.w);
    o.z = bfpack(b.x, b.y); o.w = bfpack(b.z, b.w);
    *(uint4*)((unsigned short*)out + i8) = o;
}

// f32 [K][N] -> bf16 [N][K] tiled transpose (32k x 64n per block)
__global__ __launch_bounds__(256) void convert_tr(
    const float* __restrict__ in, __bf16* __restrict__ out, int K, int N) {
    __shared__ float Ls[32][65];
    const int tid = threadIdx.x;
    const int k0 = blockIdx.y * 32;
    const int n0 = blockIdx.x * 64;
#pragma unroll
    for (int i = 0; i < 8; ++i) {
        const int idx = tid + i * 256;
        const int kr = idx >> 6, nc = idx & 63;
        Ls[kr][nc] = in[(size_t)(k0 + kr) * N + n0 + nc];
    }
    __syncthreads();
    unsigned* outu = (unsigned*)out;
#pragma unroll
    for (int i = 0; i < 4; ++i) {
        const int idx = tid + i * 256;
        const int nr = idx >> 4, ku = idx & 15;
        outu[(((size_t)(n0 + nr) * K) >> 1) + (k0 >> 1) + ku] =
            bfpack(Ls[2 * ku][nr], Ls[2 * ku + 1][nr]);
    }
}

// ===========================================================================
// WMMA GEMM:  C[M,N] = epi( A[M,K] @ Wt[N,K]^T + bias[N] )
// A: bf16 row-major.  Wt: bf16 transposed (n-major, k contiguous).
// Block: 256 threads (8 waves). Block tile 256(M) x 64(N), K-step 32.
// Each wave computes a 32x64 strip = eight 16x16 f32 accumulators.
// Async-copy double buffering, 2x-unrolled so buffer parity is static.
// EPI: 0 = f32 store, 1 = bf16 store, 2 = relu bf16 store,
//      3 = scale[row]*val f32 store, 4 = += scale[row]*val f32
// ===========================================================================
template <int EPI>
__global__ __launch_bounds__(256) void gemm_bf16_wmma(
    const __bf16* __restrict__ A, const __bf16* __restrict__ Wt,
    const float* __restrict__ bias, float* __restrict__ C,
    __bf16* __restrict__ Cbf, const float* __restrict__ scale,
    int M, int K, int N) {

    // As[buf][row][p/2]: 32 bf16 per row, A-fragment swizzled order
    __shared__ __align__(16) unsigned As[2][256][16];
    // Bs[buf][n][k/2]: n-major, k contiguous (natural B-fragment order)
    __shared__ __align__(16) unsigned Bs[2][64][16];

    const int tid  = threadIdx.x;
    const int wave = tid >> 5;
    const int lane = tid & 31;
    const int half = lane >> 4;
    const int nidx = lane & 15;
    const int m0 = blockIdx.y * 256;
    const int n0 = blockIdx.x * 64;

    // per-thread staging assignment: 2 A slots (16 halves each) + 1 B quad
    const int rowA = tid >> 1;            // slot0 row (slot1 = rowA + 128)
    const int cbA  = (tid & 1) * 16;
    const int uiA0 = aswz(cbA) >> 1;      // uint4-aligned swizzle groups
    const int uiA1 = aswz(cbA + 8) >> 1;
    const int nB   = tid >> 2;
    const int kqB  = (tid & 3) * 8;

    const __bf16* arow0 = A + (size_t)(m0 + rowA) * K + cbA;
    const __bf16* arow1 = A + (size_t)(m0 + rowA + 128) * K + cbA;
    const __bf16* brow  = Wt + (size_t)(n0 + nB) * K + kqB;

    // issue the 5 async b128 copies for the tile at k0 into buffer `buf`
    auto issue_tile = [&](int k0, int buf) {
        async_b128((unsigned)(size_t)&As[buf][rowA][uiA0],       arow0 + k0);
        async_b128((unsigned)(size_t)&As[buf][rowA][uiA1],       arow0 + k0 + 8);
        async_b128((unsigned)(size_t)&As[buf][rowA + 128][uiA0], arow1 + k0);
        async_b128((unsigned)(size_t)&As[buf][rowA + 128][uiA1], arow1 + k0 + 8);
        async_b128((unsigned)(size_t)&Bs[buf][nB][kqB >> 1],     brow + k0);
    };

    v8f acc[2][4] = {};
    const int row0 = wave * 32 + nidx;

    // consume one 32-wide K tile from buffer `buf` (8 WMMAs)
    auto compute_tile = [&](int buf) {
        Frag16 fa0, fa1;
        fa0.q.lo = *(const uint4*)&As[buf][row0][half * 8];
        fa0.q.hi = *(const uint4*)&As[buf][row0][half * 8 + 4];
        fa1.q.lo = *(const uint4*)&As[buf][row0 + 16][half * 8];
        fa1.q.hi = *(const uint4*)&As[buf][row0 + 16][half * 8 + 4];
#pragma unroll
        for (int f = 0; f < 4; ++f) {
            Frag16 fb;
            fb.q.lo = *(const uint4*)&Bs[buf][f * 16 + nidx][half * 4];
            fb.q.hi = *(const uint4*)&Bs[buf][f * 16 + nidx][half * 4 + 8];
            acc[0][f] = __builtin_amdgcn_wmma_f32_16x16x32_bf16(
                false, fa0.v, false, fb.v, (short)0, acc[0][f], false, false);
            acc[1][f] = __builtin_amdgcn_wmma_f32_16x16x32_bf16(
                false, fa1.v, false, fb.v, (short)0, acc[1][f], false, false);
        }
    };

    // software pipeline, 2x unrolled: buffer parity is compile-time constant.
    // K/32 is even for all GEMMs here (K = 512 or 2048).
    issue_tile(0, 0);                                   // 5 outstanding
    int k0 = 0;
    for (; k0 + 64 < K; k0 += 64) {
        issue_tile(k0 + 32, 1);                         // 10 outstanding
        asm volatile("s_wait_asynccnt 5" ::: "memory"); // tile in buf0 landed
        __syncthreads();
        compute_tile(0);
        __syncthreads();
        issue_tile(k0 + 64, 0);
        asm volatile("s_wait_asynccnt 5" ::: "memory"); // tile in buf1 landed
        __syncthreads();
        compute_tile(1);
        __syncthreads();
    }
    // final pair (no further prefetch after the first issue)
    issue_tile(k0 + 32, 1);
    asm volatile("s_wait_asynccnt 5" ::: "memory");
    __syncthreads();
    compute_tile(0);
    __syncthreads();
    asm volatile("s_wait_asynccnt 0" ::: "memory");
    __syncthreads();
    compute_tile(1);

    // epilogue: D layout m = r + 8*half, n = lane&15
#pragma unroll
    for (int sub = 0; sub < 2; ++sub) {
#pragma unroll
        for (int f = 0; f < 4; ++f) {
#pragma unroll
            for (int r = 0; r < 8; ++r) {
                const int row = m0 + wave * 32 + sub * 16 + r + half * 8;
                const int col = n0 + f * 16 + nidx;
                const float val = acc[sub][f][r] + bias[col];
                const size_t idx = (size_t)row * N + col;
                if (EPI == 0)      { C[idx] = val; }
                else if (EPI == 1) { Cbf[idx] = f2bf(val); }
                else if (EPI == 2) { Cbf[idx] = f2bf(fmaxf(val, 0.0f)); }
                else if (EPI == 3) { C[idx] = scale[(size_t)row * E_] * val; }
                else               { C[idx] += scale[(size_t)row * E_] * val; }
            }
        }
    }
}

// ===========================================================================
// Attention: one wave per (b, h, 16-query tile). Exact 2-pass softmax.
// q/k/v are bf16 (written by the projection GEMMs); fragment pairs (k,k+1)
// are contiguous in memory, so K/Q fragments load as single dwords.
// Output written as bf16 (feeds the O-projection GEMM).
// ===========================================================================
template <int SK, bool CAUSAL>
__global__ __launch_bounds__(32) void attn_wmma(
    const __bf16* __restrict__ q, const __bf16* __restrict__ k,
    const __bf16* __restrict__ v, __bf16* __restrict__ o) {

    __shared__ float sc[16][SK];
    __shared__ __align__(16) unsigned pb[16][SK / 2];  // packed bf16 P, swizzled

    const int lane = threadIdx.x;
    const int half = lane >> 4;
    const int nidx = lane & 15;
    const int m0 = blockIdx.x * 16;
    const int h  = blockIdx.y;
    const int b  = blockIdx.z;

    // Q fragments: two 16x32 chunks over DH=64 (dword gathers of bf16 pairs)
    FragU aQ[2];
#pragma unroll
    for (int c = 0; c < 2; ++c) {
#pragma unroll
        for (int j = 0; j < 8; ++j) {
            const int kk = c * 32 + (j >> 2) * 16 + (j & 3) * 2 + half * 8;
            const size_t base = ((size_t)(b * S_ + m0 + nidx)) * D_ + h * DH_ + kk;
            aQ[c].u[j] = *(const unsigned*)(q + base);
        }
    }

    // scores = (Q @ K^T) / sqrt(DH), with causal mask
    for (int nt = 0; nt < SK / 16; ++nt) {
        v8f s = {};
#pragma unroll
        for (int c = 0; c < 2; ++c) {
            FragU bK;
#pragma unroll
            for (int j = 0; j < 8; ++j) {
                const int kk = c * 32 + (j >> 2) * 16 + (j & 3) * 2 + half * 8;
                const size_t kb = ((size_t)(b * SK + nt * 16 + nidx)) * D_ + h * DH_ + kk;
                bK.u[j] = *(const unsigned*)(k + kb);
            }
            s = __builtin_amdgcn_wmma_f32_16x16x32_bf16(
                false, aQ[c].v, false, bK.v, (short)0, s, false, false);
        }
#pragma unroll
        for (int r = 0; r < 8; ++r) {
            const int m = r + half * 8;
            const int n = nt * 16 + nidx;
            float val = s[r] * 0.125f;  // 1/sqrt(64)
            if (CAUSAL && n > m0 + m) val = -1e9f;
            sc[m][n] = val;
        }
    }
    __syncthreads();

    // softmax over SK per row; 2 lanes per row, combine via shfl_xor(16);
    // normalized P written packed+swizzled into pb
    {
        const int rr = lane & 15;
        const int c0 = half * (SK / 2);
        float mx = -1e30f;
        for (int c = 0; c < SK / 2; ++c) mx = fmaxf(mx, sc[rr][c0 + c]);
        mx = fmaxf(mx, __shfl_xor(mx, 16, 32));
        float sum = 0.0f;
        for (int c = 0; c < SK / 2; ++c) sum += __expf(sc[rr][c0 + c] - mx);
        sum += __shfl_xor(sum, 16, 32);
        const float inv = 1.0f / sum;
        for (int c = 0; c < SK / 2; c += 2) {
            const float p0 = __expf(sc[rr][c0 + c] - mx) * inv;
            const float p1 = __expf(sc[rr][c0 + c + 1] - mx) * inv;
            const int col = c0 + c;
            const int kt  = col >> 5;
            const int kl  = col & 31;
            pb[rr][kt * 16 + (aswz(kl) >> 1)] = bfpack(p0, p1);
        }
    }
    __syncthreads();

    // O = P @ V
    const unsigned short* vu = (const unsigned short*)v;
    v8f oa[4] = {};
    for (int kt = 0; kt < SK / 32; ++kt) {
        Frag16 fp;
        fp.q.lo = *(const uint4*)&pb[nidx][kt * 16 + half * 8];
        fp.q.hi = *(const uint4*)&pb[nidx][kt * 16 + half * 8 + 4];
#pragma unroll
        for (int f = 0; f < 4; ++f) {
            FragU bV;
#pragma unroll
            for (int j = 0; j < 8; ++j) {
                const int kk = (j >> 2) * 16 + (j & 3) * 2 + half * 8;
                const size_t vb = ((size_t)(b * SK + kt * 32 + kk)) * D_ + h * DH_ + f * 16 + nidx;
                const unsigned lo = vu[vb];
                const unsigned hi = vu[vb + D_];   // next kv row
                bV.u[j] = lo | (hi << 16);
            }
            oa[f] = __builtin_amdgcn_wmma_f32_16x16x32_bf16(
                false, fp.v, false, bV.v, (short)0, oa[f], false, false);
        }
    }
#pragma unroll
    for (int f = 0; f < 4; ++f)
#pragma unroll
        for (int r = 0; r < 8; ++r) {
            const int m = r + half * 8;
            o[((size_t)(b * S_ + m0 + m)) * D_ + h * DH_ + f * 16 + nidx] = f2bf(oa[f][r]);
        }
}

// ===========================================================================
// Fused residual add + LayerNorm; optional bf16 copy for the next GEMM.
// ===========================================================================
template <bool WBF>
__global__ __launch_bounds__(256) void add_ln(
    const float* __restrict__ a, const float* __restrict__ r,
    const float* __restrict__ g, const float* __restrict__ be,
    float* __restrict__ out, __bf16* __restrict__ outbf) {
    const int row  = blockIdx.x * 8 + (threadIdx.x >> 5);
    const int lane = threadIdx.x & 31;
    float vb[16];
    float sum = 0.0f, sq = 0.0f;
#pragma unroll
    for (int i = 0; i < 16; ++i) {
        const int col = lane + i * 32;
        const float t = a[(size_t)row * D_ + col] + r[(size_t)row * D_ + col];
        vb[i] = t; sum += t; sq += t * t;
    }
#pragma unroll
    for (int off = 16; off >= 1; off >>= 1) {
        sum += __shfl_xor(sum, off, 32);
        sq  += __shfl_xor(sq,  off, 32);
    }
    const float mean = sum * (1.0f / D_);
    const float var  = sq * (1.0f / D_) - mean * mean;
    const float rstd = rsqrtf(var + EPS_);
#pragma unroll
    for (int i = 0; i < 16; ++i) {
        const int col = lane + i * 32;
        const float y = (vb[i] - mean) * rstd * g[col] + be[col];
        out[(size_t)row * D_ + col] = y;
        if (WBF) outbf[(size_t)row * D_ + col] = f2bf(y);
    }
}

// ===========================================================================
// MoE gate: per-row softmax over E=4 logits + importance accumulation.
// ===========================================================================
__global__ __launch_bounds__(256) void gate_softmax(
    const float* __restrict__ x, const float* __restrict__ gw,
    const float* __restrict__ gb, float* __restrict__ gates,
    float* __restrict__ auxsum) {
    const int row  = blockIdx.x * 8 + (threadIdx.x >> 5);
    const int lane = threadIdx.x & 31;
    float acc[E_] = {};
    for (int i = lane; i < D_; i += 32) {
        const float xv = x[(size_t)row * D_ + i];
        const float4 w = *(const float4*)&gw[(size_t)i * E_];
        acc[0] += xv * w.x; acc[1] += xv * w.y;
        acc[2] += xv * w.z; acc[3] += xv * w.w;
    }
#pragma unroll
    for (int off = 16; off >= 1; off >>= 1)
#pragma unroll
        for (int e = 0; e < E_; ++e) acc[e] += __shfl_xor(acc[e], off, 32);
    if (lane == 0) {
        float mx = -1e30f;
#pragma unroll
        for (int e = 0; e < E_; ++e) { acc[e] += gb[e]; mx = fmaxf(mx, acc[e]); }
        float s = 0.0f;
#pragma unroll
        for (int e = 0; e < E_; ++e) { acc[e] = __expf(acc[e] - mx); s += acc[e]; }
        const float inv = 1.0f / s;
#pragma unroll
        for (int e = 0; e < E_; ++e) {
            const float ge = acc[e] * inv;
            gates[(size_t)row * E_ + e] = ge;
            atomicAdd(&auxsum[e], ge);
        }
    }
}

__global__ void zero_aux(float* auxsum) { auxsum[threadIdx.x] = 0.0f; }

__global__ void aux_final(const float* __restrict__ auxsum, float* __restrict__ out) {
    float s = 0.0f;
#pragma unroll
    for (int e = 0; e < E_; ++e) {
        const float m = auxsum[e] * (1.0f / (float)BS_);
        s += m * m;
    }
    out[0] = (float)E_ * s;
}

// ===========================================================================
// Host launch
// ===========================================================================
extern "C" void kernel_launch(void* const* d_in, const int* in_sizes, int n_in,
                              void* d_out, int out_size, void* d_ws, size_t ws_size,
                              hipStream_t stream) {
    const float* x      = (const float*)d_in[0];
    const float* cross  = (const float*)d_in[1];
    const float* aw[8]  = { (const float*)d_in[2],  (const float*)d_in[4],
                            (const float*)d_in[6],  (const float*)d_in[8],
                            (const float*)d_in[10], (const float*)d_in[12],
                            (const float*)d_in[14], (const float*)d_in[16] };
    const float* sa_bq = (const float*)d_in[3];   const float* sa_bk = (const float*)d_in[5];
    const float* sa_bv = (const float*)d_in[7];   const float* sa_bo = (const float*)d_in[9];
    const float* ca_bq = (const float*)d_in[11];  const float* ca_bk = (const float*)d_in[13];
    const float* ca_bv = (const float*)d_in[15];  const float* ca_bo = (const float*)d_in[17];
    const float* gate_w = (const float*)d_in[18]; const float* gate_b = (const float*)d_in[19];
    const float* exp_w1 = (const float*)d_in[20]; const float* exp_b1 = (const float*)d_in[21];
    const float* exp_w2 = (const float*)d_in[22]; const float* exp_b2 = (const float*)d_in[23];
    const float* n1_g = (const float*)d_in[24];   const float* n1_b = (const float*)d_in[25];
    const float* n2_g = (const float*)d_in[26];   const float* n2_b = (const float*)d_in[27];
    const float* n3_g = (const float*)d_in[28];   const float* n3_b = (const float*)d_in[29];

    float* out = (float*)d_out;                 // [BS*D] + 1 aux scalar
    float* ws  = (float*)d_ws;

    const size_t CH = (size_t)BS_ * D_;         // 8M elements
    // f32 region
    float* tmp_b  = ws;                         // o-proj out; phase3: MoE y
    float* x1_buf = ws + CH;
    float* x2_buf = ws + 2 * CH;
    float* gates  = ws + 3 * CH;                // BS*E
    float* auxsum = gates + (size_t)BS_ * E_;   // E floats
    float* y_buf  = tmp_b;
    // bf16 region
    __bf16* bfb     = (__bf16*)(ws + 3 * CH + 131072);
    __bf16* xbf     = bfb;
    __bf16* crossbf = bfb + CH;                 // BT*D used
    __bf16* qbf     = bfb + 2 * CH;             // phase3: hbf overlays [2CH,6CH)
    __bf16* kbf     = bfb + 3 * CH;
    __bf16* vbf     = bfb + 4 * CH;
    __bf16* attnbf  = bfb + 5 * CH;
    __bf16* x1bf    = bfb + 6 * CH;
    __bf16* x2bf    = bfb + 7 * CH;
    __bf16* awt     = bfb + 8 * CH;             // 8 transposed attn weights
    __bf16* w1t     = awt + (size_t)8 * D_ * D_;
    __bf16* w2t     = w1t + (size_t)E_ * D_ * DF_;
    __bf16* hbf     = qbf;                      // [BS, DF] bf16

    const dim3 blk256(256);

    // ---- one-time conversions (bandwidth-bound) ----
    convert_rm<<<CH / 2048, blk256, 0, stream>>>(x, xbf);
    convert_rm<<<((size_t)BT_ * D_) / 2048, blk256, 0, stream>>>(cross, crossbf);
    for (int i = 0; i < 8; ++i)
        convert_tr<<<dim3(D_ / 64, D_ / 32), blk256, 0, stream>>>(
            aw[i], awt + (size_t)i * D_ * D_, D_, D_);
    for (int e = 0; e < E_; ++e) {
        convert_tr<<<dim3(DF_ / 64, D_ / 32), blk256, 0, stream>>>(
            exp_w1 + (size_t)e * D_ * DF_, w1t + (size_t)e * D_ * DF_, D_, DF_);
        convert_tr<<<dim3(D_ / 64, DF_ / 32), blk256, 0, stream>>>(
            exp_w2 + (size_t)e * DF_ * D_, w2t + (size_t)e * DF_ * D_, DF_, D_);
    }
    const __bf16* sa_wq_t = awt;
    const __bf16* sa_wk_t = awt + (size_t)1 * D_ * D_;
    const __bf16* sa_wv_t = awt + (size_t)2 * D_ * D_;
    const __bf16* sa_wo_t = awt + (size_t)3 * D_ * D_;
    const __bf16* ca_wq_t = awt + (size_t)4 * D_ * D_;
    const __bf16* ca_wk_t = awt + (size_t)5 * D_ * D_;
    const __bf16* ca_wv_t = awt + (size_t)6 * D_ * D_;
    const __bf16* ca_wo_t = awt + (size_t)7 * D_ * D_;

    const dim3 gProj(D_ / 64, BS_ / 256);
    const dim3 gProjT(D_ / 64, BT_ / 256);
    const dim3 gUp(DF_ / 64, BS_ / 256);
    const dim3 gDn(D_ / 64, BS_ / 256);

    // ---- block 1: causal self-attention ----
    gemm_bf16_wmma<1><<<gProj, blk256, 0, stream>>>(xbf, sa_wq_t, sa_bq, nullptr, qbf, nullptr, BS_, D_, D_);
    gemm_bf16_wmma<1><<<gProj, blk256, 0, stream>>>(xbf, sa_wk_t, sa_bk, nullptr, kbf, nullptr, BS_, D_, D_);
    gemm_bf16_wmma<1><<<gProj, blk256, 0, stream>>>(xbf, sa_wv_t, sa_bv, nullptr, vbf, nullptr, BS_, D_, D_);
    attn_wmma<S_, true><<<dim3(S_ / 16, H_, B_), dim3(32), 0, stream>>>(qbf, kbf, vbf, attnbf);
    gemm_bf16_wmma<0><<<gProj, blk256, 0, stream>>>(attnbf, sa_wo_t, sa_bo, tmp_b, nullptr, nullptr, BS_, D_, D_);
    add_ln<true><<<BS_ / 8, blk256, 0, stream>>>(x, tmp_b, n1_g, n1_b, x1_buf, x1bf);

    // ---- block 2: cross-attention ----
    gemm_bf16_wmma<1><<<gProj,  blk256, 0, stream>>>(x1bf, ca_wq_t, ca_bq, nullptr, qbf, nullptr, BS_, D_, D_);
    gemm_bf16_wmma<1><<<gProjT, blk256, 0, stream>>>(crossbf, ca_wk_t, ca_bk, nullptr, kbf, nullptr, BT_, D_, D_);
    gemm_bf16_wmma<1><<<gProjT, blk256, 0, stream>>>(crossbf, ca_wv_t, ca_bv, nullptr, vbf, nullptr, BT_, D_, D_);
    attn_wmma<T_, false><<<dim3(S_ / 16, H_, B_), dim3(32), 0, stream>>>(qbf, kbf, vbf, attnbf);
    gemm_bf16_wmma<0><<<gProj, blk256, 0, stream>>>(attnbf, ca_wo_t, ca_bo, tmp_b, nullptr, nullptr, BS_, D_, D_);
    add_ln<true><<<BS_ / 8, blk256, 0, stream>>>(x1_buf, tmp_b, n2_g, n2_b, x2_buf, x2bf);

    // ---- block 3: MoE FFN ----
    zero_aux<<<1, E_, 0, stream>>>(auxsum);
    gate_softmax<<<BS_ / 8, blk256, 0, stream>>>(x2_buf, gate_w, gate_b, gates, auxsum);
    for (int e = 0; e < E_; ++e) {
        const __bf16* w1 = w1t + (size_t)e * D_ * DF_;
        const float*  b1 = exp_b1 + (size_t)e * DF_;
        const __bf16* w2 = w2t + (size_t)e * DF_ * D_;
        const float*  b2 = exp_b2 + (size_t)e * D_;
        gemm_bf16_wmma<2><<<gUp, blk256, 0, stream>>>(x2bf, w1, b1, nullptr, hbf, nullptr, BS_, D_, DF_);
        if (e == 0)
            gemm_bf16_wmma<3><<<gDn, blk256, 0, stream>>>(hbf, w2, b2, y_buf, nullptr, gates + e, BS_, DF_, D_);
        else
            gemm_bf16_wmma<4><<<gDn, blk256, 0, stream>>>(hbf, w2, b2, y_buf, nullptr, gates + e, BS_, DF_, D_);
    }
    add_ln<false><<<BS_ / 8, blk256, 0, stream>>>(x2_buf, y_buf, n3_g, n3_b, out, nullptr);
    aux_final<<<1, 1, 0, stream>>>(auxsum, out + (size_t)BS_ * D_);
}
